// LIONLinearAttentionLayer_74062416052690
// MI455X (gfx1250) — matrix operand order repaired and checked by reading, not compiled, not run on hardware
//
#include <hip/hip_runtime.h>
#include <hip/hip_bf16.h>
#include <cstdint>

// ---------------------------------------------------------------------------
// Problem constants (from reference): B=2, T=2048, D=1024, H=16, FFN=4096, K=64
// ---------------------------------------------------------------------------
#define BB 2
#define TT 2048
#define DD 1024
#define HH 16
#define FF 4096
#define KDIM 64
#define KV 65          // K + 1 (denominator channel)
#define CH 256         // scan chunk length
#define NC (TT / CH)   // chunks per sequence (8)
#define NT 4           // n-tiles (16 cols) per wave in the GEMM

typedef __bf16 bf16;
typedef __bf16 v8bf  __attribute__((ext_vector_type(8)));
typedef __bf16 v16bf __attribute__((ext_vector_type(16)));
typedef float  v8f   __attribute__((ext_vector_type(8)));

// ---------------------------------------------------------------------------
// Weight staging: fp32 [Kd x N] row-major -> bf16 pre-swizzled into the exact
// WMMA B-fragment layout (ISA 7.12.2, 16-bit B 32x16):
//   tile (kt, nt) of 32(K) x 16(N); lane l holds 16 contiguous bf16:
//     value i = W[kt*32 + (i<8 ? (l>=16)*8 + i : 16 + (l>=16)*8 + i-8)][nt*16 + (l&15)]
//   packed[( (nt*KT + kt)*32 + l )*16 + i]
// So a 32x16 tile is 1 KB contiguous -> perfect for async b128 copies to LDS.
// ---------------------------------------------------------------------------
__global__ void pack_w_bf16_kernel(const float* __restrict__ W,
                                   bf16* __restrict__ out, int Kd, int N) {
    const int KT = Kd >> 5;
    const size_t total = (size_t)Kd * N;
    size_t idx = (size_t)blockIdx.x * blockDim.x + threadIdx.x;
    const size_t stride = (size_t)gridDim.x * blockDim.x;
    for (; idx < total; idx += stride) {
        const int i = (int)(idx & 15);
        const int l = (int)((idx >> 4) & 31);
        const size_t tile = idx >> 9;
        const int kt = (int)(tile % KT);
        const int nt = (int)(tile / KT);
        const int krow = kt * 32 + ((i < 8) ? ((l >> 4) * 8 + i) : (16 + (l >> 4) * 8 + (i - 8)));
        const int col  = nt * 16 + (l & 15);
        out[idx] = (bf16)W[(size_t)krow * N + col];
    }
}

// ---------------------------------------------------------------------------
// LayerNorm over D, output bf16 (feeds WMMA GEMMs). One block per row.
// ---------------------------------------------------------------------------
__global__ __launch_bounds__(256)
void ln_to_bf16_kernel(const float* __restrict__ x, const float* __restrict__ w,
                       const float* __restrict__ b, bf16* __restrict__ out, int Dd) {
    __shared__ float s_sum[256];
    __shared__ float s_sq[256];
    const int row = blockIdx.x;
    const float* xr = x + (size_t)row * Dd;
    float s = 0.f, sq = 0.f;
    for (int d = threadIdx.x; d < Dd; d += 256) {
        float v = xr[d]; s += v; sq += v * v;
    }
    s_sum[threadIdx.x] = s; s_sq[threadIdx.x] = sq;
    __syncthreads();
    for (int st = 128; st > 0; st >>= 1) {
        if (threadIdx.x < st) {
            s_sum[threadIdx.x] += s_sum[threadIdx.x + st];
            s_sq[threadIdx.x]  += s_sq[threadIdx.x + st];
        }
        __syncthreads();
    }
    const float mean = s_sum[0] / Dd;
    const float var  = s_sq[0] / Dd - mean * mean;
    const float rstd = rsqrtf(var + 1e-5f);
    for (int d = threadIdx.x; d < Dd; d += 256) {
        float v = (xr[d] - mean) * rstd * w[d] + b[d];
        out[(size_t)row * Dd + d] = (bf16)v;
    }
}

// ---------------------------------------------------------------------------
// bf16 WMMA GEMM: C = act(A[M,Kd] x W[Kd,N] + bias) (+ resid), fp32 accum.
// Block = 8 waves; wave tile = 16 rows x 64 cols; block tile = 128 x 64.
//
// B path (the 8x-redundant one) goes through LDS via CDNA5 async copies:
// per K-step the block's 4 KB of B fragments are staged ONCE with
// global_load_async_to_lds_b128 (1 x 16B transfer per thread), double-
// buffered and tracked with ASYNCcnt (s_wait_asynccnt), overlapping the
// next tile's copy with the current tile's WMMAs. Waves then read the
// fragments with ds_load_b128 (conflict-light: 32B/lane contiguous).
// A path: per-lane contiguous b128 global loads (no intra-block redundancy).
// act: 0 none, 1 elu+1 (phi), 2 exact gelu. Stores fp32 and/or bf16.
// ---------------------------------------------------------------------------
__global__ __launch_bounds__(256)
void wmma_gemm_kernel(const bf16* __restrict__ A, const bf16* __restrict__ Bp,
                      const float* __restrict__ bias, const float* __restrict__ resid,
                      float* __restrict__ outF, bf16* __restrict__ outB,
                      int M, int N, int Kd, int act) {
    // 2 buffers x 4 n-tiles x 512 bf16 (1 KB/tile) = 8 KB LDS
    __shared__ __align__(32) bf16 sB[2][NT * 512];

    const int tid  = threadIdx.x;
    const int wave = tid >> 5;
    const int lane = tid & 31;
    const int m_base = blockIdx.x * 128 + wave * 16;
    const int n0 = blockIdx.y * (16 * NT);
    const int lhalf = (lane >> 4) & 1;
    const int mrow = m_base + (lane & 15);
    const int KT = Kd >> 5;
    const int nt0 = n0 >> 4;

    // Async stage helper indices: thread -> (tile j, 16B chunk sub)
    const int sj   = tid >> 6;     // 0..3
    const int ssub = tid & 63;     // 64 x 16B chunks per 1 KB tile

    // Stage kt=0 into buffer 0 (one async b128 per thread = 4 KB per block)
    {
        const bf16* src = Bp + ((size_t)(nt0 + sj) * KT + 0) * 512 + ssub * 8;
        const uint32_t lds = (uint32_t)(uintptr_t)(&sB[0][sj * 512 + ssub * 8]);
        asm volatile("global_load_async_to_lds_b128 %0, %1, off"
                     :: "v"(lds), "v"((uint64_t)(uintptr_t)src) : "memory");
    }

    v8f cacc[NT] = {};
    const bf16* pa_row = A + (size_t)mrow * Kd + lhalf * 8;

    for (int kt = 0; kt < KT; ++kt) {
        const int cur = kt & 1;
        // Prefetch next K-step's B tiles into the other buffer, then wait for
        // the CURRENT buffer (async loads complete in order -> cnt<=1 is ok).
        if (kt + 1 < KT) {
            const bf16* src = Bp + ((size_t)(nt0 + sj) * KT + (kt + 1)) * 512 + ssub * 8;
            const uint32_t lds = (uint32_t)(uintptr_t)(&sB[cur ^ 1][sj * 512 + ssub * 8]);
            asm volatile("global_load_async_to_lds_b128 %0, %1, off"
                         :: "v"(lds), "v"((uint64_t)(uintptr_t)src) : "memory");
            asm volatile("s_wait_asynccnt 0x1" ::: "memory");
        } else {
            asm volatile("s_wait_asynccnt 0x0" ::: "memory");
        }
        __syncthreads();   // all waves' copies for `cur` visible

        // A 16x32 bf16 fragment: two contiguous 16B groups per lane
        const bf16* pa = pa_row + kt * 32;
        const v8bf alo = *(const v8bf*)pa;
        const v8bf ahi = *(const v8bf*)(pa + 16);
        v16bf a;
        #pragma unroll
        for (int i = 0; i < 8; ++i) { a[i] = alo[i]; a[i + 8] = ahi[i]; }

        // B fragments from LDS: 32 contiguous bytes per lane per tile
        const bf16* sb = &sB[cur][lane * 16];
        const v16bf b0 = *(const v16bf*)(sb);
        const v16bf b1 = *(const v16bf*)(sb + 512);
        const v16bf b2 = *(const v16bf*)(sb + 1024);
        const v16bf b3 = *(const v16bf*)(sb + 1536);
        cacc[0] = __builtin_amdgcn_wmma_f32_16x16x32_bf16(false, a, false, b0, (short)0, cacc[0], false, false);
        cacc[1] = __builtin_amdgcn_wmma_f32_16x16x32_bf16(false, a, false, b1, (short)0, cacc[1], false, false);
        cacc[2] = __builtin_amdgcn_wmma_f32_16x16x32_bf16(false, a, false, b2, (short)0, cacc[2], false, false);
        cacc[3] = __builtin_amdgcn_wmma_f32_16x16x32_bf16(false, a, false, b3, (short)0, cacc[3], false, false);

        __syncthreads();   // everyone done reading `cur` before it is re-staged
    }

    if (m_base >= M) return;

    // Epilogue: C/D VGPR r -> M = m_base + r + 8*lhalf, N = n0 + j*16 + (lane&15)
    #pragma unroll
    for (int j = 0; j < NT; ++j) {
        const int col = n0 + j * 16 + (lane & 15);
        #pragma unroll
        for (int r = 0; r < 8; ++r) {
            const int m = m_base + r + lhalf * 8;
            float v = cacc[j][r];
            if (bias) v += bias[col];
            if (act == 1)      v = (v > 0.f) ? (v + 1.f) : __expf(v);           // elu+1
            else if (act == 2) v = 0.5f * v * (1.f + erff(v * 0.70710678118654752f));
            const size_t o = (size_t)m * N + col;
            if (resid) v += resid[o];
            if (outF) outF[o] = v;
            if (outB) outB[o] = (bf16)v;
        }
    }
}

// ---------------------------------------------------------------------------
// Decay gate: lam[b*T+t, h] = max(sigmoid(xn . dec_w[:,h] + dec_b[h]), 1e-8)
// ---------------------------------------------------------------------------
__global__ __launch_bounds__(256)
void lam_kernel(const bf16* __restrict__ xn, const float* __restrict__ dec_w,
                const float* __restrict__ dec_b, float* __restrict__ lam) {
    __shared__ float red[256];
    const int row  = blockIdx.x;           // b*T + t
    const int h    = threadIdx.x >> 4;     // 0..15
    const int part = threadIdx.x & 15;
    float acc = 0.f;
    for (int d = part; d < DD; d += 16)
        acc += (float)xn[(size_t)row * DD + d] * dec_w[d * HH + h];
    red[threadIdx.x] = acc;
    __syncthreads();
    if (part == 0) {
        float s = 0.f;
        #pragma unroll
        for (int i = 0; i < 16; ++i) s += red[(h << 4) + i];
        s += dec_b[h];
        float sg = 1.f / (1.f + __expf(-s));
        lam[(size_t)row * HH + h] = fmaxf(sg, 1e-8f);
    }
}

// ---------------------------------------------------------------------------
// Chunked bidirectional selective-decay scan (decay is SCALAR per (b,h,t), so
// cross-chunk state composes as  S = S_local + (prod lam) * S_carry).
// Pass 1: per-chunk local scans (512 parallel blocks).
// ---------------------------------------------------------------------------
__global__ __launch_bounds__(96)
void scan_chunk_kernel(const float* __restrict__ phi_q, const float* __restrict__ phi_k,
                       const float* __restrict__ vp, const float* __restrict__ lam,
                       float* __restrict__ out_f, float* __restrict__ out_b,
                       float* __restrict__ Sbuf, float* __restrict__ pprod,
                       float* __restrict__ ptot) {
    __shared__ float sq[KDIM];
    __shared__ float sk[KDIM];
    __shared__ float slam[1];
    const int bh = blockIdx.x, c = blockIdx.y, dir = blockIdx.z;
    const int b = bh / HH, h = bh % HH;
    const int tid = threadIdx.x;
    float* __restrict__ outp = dir ? out_b : out_f;
    const size_t pbase = ((size_t)(dir * BB * HH + bh)) * TT;

    float S[KDIM];
    #pragma unroll
    for (int k = 0; k < KDIM; ++k) S[k] = 0.f;
    float p = 1.f;

    for (int step = 0; step < CH; ++step) {
        const int t = dir ? (c * CH + (CH - 1 - step)) : (c * CH + step);
        const size_t rb = ((size_t)(b * TT + t)) * DD + h * KDIM;
        if (tid < KDIM) { sq[tid] = phi_q[rb + tid]; sk[tid] = phi_k[rb + tid]; }
        if (tid == 0)   slam[0] = lam[(size_t)(b * TT + t) * HH + h];
        __syncthreads();
        const float l = slam[0];
        const float vcol = (tid < KDIM) ? vp[rb + tid] : 1.f;
        float acc = 0.f;
        float pstore;
        if (dir == 0) {               // fwd: S = l*S + k(x)v ; out = q.S
            p *= l;                   // P_t includes lam_t
            pstore = p;
            #pragma unroll
            for (int k = 0; k < KDIM; ++k) {
                S[k] = S[k] * l + sk[k] * vcol;
                acc += sq[k] * S[k];
            }
        } else {                      // bwd: out = q.R (pre) ; R = l*(k(x)v + R)
            pstore = p;               // Q_{t+1}
            #pragma unroll
            for (int k = 0; k < KDIM; ++k) acc += sq[k] * S[k];
            #pragma unroll
            for (int k = 0; k < KDIM; ++k) S[k] = l * (sk[k] * vcol + S[k]);
            p *= l;
        }
        if (tid < KV) outp[((size_t)bh * TT + t) * KV + tid] = acc;
        if (tid == 0) pprod[pbase + t] = pstore;
        __syncthreads();
    }
    const size_t sb = ((size_t)((dir * BB * HH + bh) * NC + c)) * (KDIM * KV);
    if (tid < KV) {
        #pragma unroll
        for (int k = 0; k < KDIM; ++k) Sbuf[sb + k * KV + tid] = S[k];
    }
    if (tid == 0) ptot[(size_t)(dir * BB * HH + bh) * NC + c] = p;
}

// ---------------------------------------------------------------------------
// Pass 2: sequential carry combine across the NC chunks, in place:
//   on exit Sbuf[c] = carry state ENTERING chunk c.
// ---------------------------------------------------------------------------
__global__ __launch_bounds__(256)
void scan_carry_kernel(float* __restrict__ Sbuf, const float* __restrict__ ptot) {
    const int bh = blockIdx.x, dir = blockIdx.y;
    const size_t base = ((size_t)(dir * BB * HH + bh)) * NC * (KDIM * KV);
    const size_t pb = (size_t)(dir * BB * HH + bh) * NC;
    for (int idx = threadIdx.x; idx < KDIM * KV; idx += 256) {
        float carry = 0.f;
        if (dir == 0) {
            for (int c = 0; c < NC; ++c) {
                const size_t o = base + (size_t)c * (KDIM * KV) + idx;
                const float tmp = Sbuf[o];
                Sbuf[o] = carry;
                carry = tmp + ptot[pb + c] * carry;
            }
        } else {
            for (int c = NC - 1; c >= 0; --c) {
                const size_t o = base + (size_t)c * (KDIM * KV) + idx;
                const float tmp = Sbuf[o];
                Sbuf[o] = carry;
                carry = tmp + ptot[pb + c] * carry;
            }
        }
    }
}

// ---------------------------------------------------------------------------
// Pass 3: cross-chunk correction, fully parallel over chunks:
//   out[t][v] += pprod[t] * sum_k q_t[k] * carry[k][v]
// ---------------------------------------------------------------------------
__global__ __launch_bounds__(96)
void scan_fixup_kernel(const float* __restrict__ phi_q, const float* __restrict__ Sbuf,
                       const float* __restrict__ pprod,
                       float* __restrict__ out_f, float* __restrict__ out_b) {
    __shared__ float sq[KDIM];
    const int bh = blockIdx.x, c = blockIdx.y, dir = blockIdx.z;
    const int b = bh / HH, h = bh % HH;
    const int tid = threadIdx.x;
    float* __restrict__ outp = dir ? out_b : out_f;
    const size_t sb = ((size_t)((dir * BB * HH + bh) * NC + c)) * (KDIM * KV);
    float Sc[KDIM];
    #pragma unroll
    for (int k = 0; k < KDIM; ++k) Sc[k] = 0.f;
    if (tid < KV) {
        #pragma unroll
        for (int k = 0; k < KDIM; ++k) Sc[k] = Sbuf[sb + k * KV + tid];
    }
    const size_t pbase = ((size_t)(dir * BB * HH + bh)) * TT;
    for (int step = 0; step < CH; ++step) {
        const int t = c * CH + step;
        const size_t rb = ((size_t)(b * TT + t)) * DD + h * KDIM;
        if (tid < KDIM) sq[tid] = phi_q[rb + tid];
        __syncthreads();
        if (tid < KV) {
            float acc = 0.f;
            #pragma unroll
            for (int k = 0; k < KDIM; ++k) acc += sq[k] * Sc[k];
            const size_t o = ((size_t)bh * TT + t) * KV + tid;
            outp[o] += pprod[pbase + t] * acc;
        }
        __syncthreads();
    }
}

// ---------------------------------------------------------------------------
// attn = (num_f+num_b) / max(den_f+den_b, 1e-6), laid out (b*T+t, h*K+k), bf16
// ---------------------------------------------------------------------------
__global__ void attn_combine_kernel(const float* __restrict__ out_f,
                                    const float* __restrict__ out_b,
                                    bf16* __restrict__ attn) {
    const size_t total = (size_t)BB * HH * TT * KDIM;
    size_t idx = (size_t)blockIdx.x * blockDim.x + threadIdx.x;
    if (idx >= total) return;
    const int k = (int)(idx % KDIM);
    size_t r = idx / KDIM;
    const int t = (int)(r % TT);
    const int bh = (int)(r / TT);
    const int b = bh / HH, h = bh % HH;
    const size_t base = ((size_t)bh * TT + t) * KV;
    const float num = out_f[base + k] + out_b[base + k];
    const float den = fmaxf(out_f[base + KDIM] + out_b[base + KDIM], 1e-6f);
    attn[((size_t)(b * TT + t)) * DD + h * KDIM + k] = (bf16)(num / den);
}

// ---------------------------------------------------------------------------
// Host-side launch. Inputs in setup_inputs() order (all fp32).
// ---------------------------------------------------------------------------
extern "C" void kernel_launch(void* const* d_in, const int* in_sizes, int n_in,
                              void* d_out, int out_size, void* d_ws, size_t ws_size,
                              hipStream_t stream) {
    (void)in_sizes; (void)n_in; (void)out_size; (void)ws_size;
    const float* x     = (const float*)d_in[0];
    const float* q_w   = (const float*)d_in[1];
    const float* q_b   = (const float*)d_in[2];
    const float* k_w   = (const float*)d_in[3];
    const float* k_b   = (const float*)d_in[4];
    const float* v_w   = (const float*)d_in[5];
    const float* v_b   = (const float*)d_in[6];
    const float* o_w   = (const float*)d_in[7];
    const float* o_b   = (const float*)d_in[8];
    const float* ln1_w = (const float*)d_in[9];
    const float* ln1_b = (const float*)d_in[10];
    const float* ln2_w = (const float*)d_in[11];
    const float* ln2_b = (const float*)d_in[12];
    const float* dec_w = (const float*)d_in[13];
    const float* dec_b = (const float*)d_in[14];
    const float* f1_w  = (const float*)d_in[15];
    const float* f1_b  = (const float*)d_in[16];
    const float* f2_w  = (const float*)d_in[17];
    const float* f2_b  = (const float*)d_in[18];
    float* out = (float*)d_out;

    const size_t MROWS = (size_t)BB * TT;       // 4096 token rows

    // Bump allocator with lifetime-based reuse (peak ~126 MB).
    char* ws = (char*)d_ws;
    size_t off = 0;
    auto alloc = [&](size_t bytes) -> void* {
        void* p = ws + off;
        off = (off + bytes + 255) & ~(size_t)255;
        return p;
    };
    bf16* wq  = (bf16*)alloc((size_t)DD * DD * 2);
    bf16* wk  = (bf16*)alloc((size_t)DD * DD * 2);
    bf16* wv  = (bf16*)alloc((size_t)DD * DD * 2);
    bf16* wo  = (bf16*)alloc((size_t)DD * DD * 2);
    bf16* wf1 = (bf16*)alloc((size_t)DD * FF * 2);
    bf16* wf2 = (bf16*)alloc((size_t)FF * DD * 2);
    bf16*  xn1   = (bf16*) alloc(MROWS * DD * 2);
    float* phi_q = (float*)alloc(MROWS * DD * 4);
    float* phi_k = (float*)alloc(MROWS * DD * 4);
    float* vproj = (float*)alloc(MROWS * DD * 4);
    float* lam   = (float*)alloc(MROWS * HH * 4);
    float* out_f = (float*)alloc((size_t)BB * HH * TT * KV * 4);
    float* out_b = (float*)alloc((size_t)BB * HH * TT * KV * 4);
    float* Sbuf  = (float*)alloc((size_t)2 * BB * HH * NC * KDIM * KV * 4);
    float* pprod = (float*)alloc((size_t)2 * BB * HH * TT * 4);
    float* ptot  = (float*)alloc((size_t)2 * BB * HH * NC * 4);
    // Reuse (lifetimes disjoint on the single stream):
    bf16*  attn = xn1;            // xn1 dead after QKV/decay GEMMs
    float* x1   = phi_q;          // phi dead after scan passes
    bf16*  xn2  = (bf16*)phi_k;
    bf16*  hbuf = (bf16*)out_f;   // out_f+out_b contiguous: 34.1MB >= 33.5MB gelu buf

    // Stage weights: bf16 + WMMA B-fragment swizzle (L2-resident: ~25 MB).
    pack_w_bf16_kernel<<<2048, 256, 0, stream>>>(q_w, wq, DD, DD);
    pack_w_bf16_kernel<<<2048, 256, 0, stream>>>(k_w, wk, DD, DD);
    pack_w_bf16_kernel<<<2048, 256, 0, stream>>>(v_w, wv, DD, DD);
    pack_w_bf16_kernel<<<2048, 256, 0, stream>>>(o_w, wo, DD, DD);
    pack_w_bf16_kernel<<<4096, 256, 0, stream>>>(f1_w, wf1, DD, FF);
    pack_w_bf16_kernel<<<4096, 256, 0, stream>>>(f2_w, wf2, FF, DD);

    // LN1 -> bf16
    ln_to_bf16_kernel<<<(unsigned)MROWS, 256, 0, stream>>>(x, ln1_w, ln1_b, xn1, DD);

    // QKV projections (phi fused for Q/K), decay gate
    dim3 gD((unsigned)(MROWS / 128), DD / (16 * NT));
    wmma_gemm_kernel<<<gD, 256, 0, stream>>>(xn1, wq, q_b, nullptr, phi_q, nullptr,
                                             (int)MROWS, DD, DD, /*act=*/1);
    wmma_gemm_kernel<<<gD, 256, 0, stream>>>(xn1, wk, k_b, nullptr, phi_k, nullptr,
                                             (int)MROWS, DD, DD, /*act=*/1);
    wmma_gemm_kernel<<<gD, 256, 0, stream>>>(xn1, wv, v_b, nullptr, vproj, nullptr,
                                             (int)MROWS, DD, DD, /*act=*/0);
    lam_kernel<<<(unsigned)MROWS, 256, 0, stream>>>(xn1, dec_w, dec_b, lam);

    // Chunked bidirectional scan: parallel local scans, carry combine, fixup.
    scan_chunk_kernel<<<dim3(BB * HH, NC, 2), 96, 0, stream>>>(
        phi_q, phi_k, vproj, lam, out_f, out_b, Sbuf, pprod, ptot);
    scan_carry_kernel<<<dim3(BB * HH, 2), 256, 0, stream>>>(Sbuf, ptot);
    scan_fixup_kernel<<<dim3(BB * HH, NC, 2), 96, 0, stream>>>(
        phi_q, Sbuf, pprod, out_f, out_b);

    // num/den combine -> bf16 attention activations
    {
        size_t tot = (size_t)BB * HH * TT * KDIM;
        attn_combine_kernel<<<(unsigned)((tot + 255) / 256), 256, 0, stream>>>(
            out_f, out_b, attn);
    }

    // O-projection + residual (fp32 out), LN2, FFN1+gelu (bf16 out), FFN2+residual
    wmma_gemm_kernel<<<gD, 256, 0, stream>>>(attn, wo, o_b, /*resid=*/x, x1, nullptr,
                                             (int)MROWS, DD, DD, /*act=*/0);
    ln_to_bf16_kernel<<<(unsigned)MROWS, 256, 0, stream>>>(x1, ln2_w, ln2_b, xn2, DD);
    dim3 gF((unsigned)(MROWS / 128), FF / (16 * NT));
    wmma_gemm_kernel<<<gF, 256, 0, stream>>>(xn2, wf1, f1_b, nullptr, nullptr, hbuf,
                                             (int)MROWS, FF, DD, /*act=*/2);
    wmma_gemm_kernel<<<gD, 256, 0, stream>>>(hbuf, wf2, f2_b, /*resid=*/x1, out, nullptr,
                                             (int)MROWS, DD, FF, /*act=*/0);
}
